// RingMemoryV7_65240553226485
// MI455X (gfx1250) — matrix operand order repaired and checked by reading, not compiled, not run on hardware
//
#include <hip/hip_runtime.h>
#include <hip/hip_bf16.h>
#include <math.h>

// Problem constants (match reference)
#define Bsz 128
#define Tsz 1024
#define Dsz 256
#define Nsz 64

typedef __attribute__((ext_vector_type(16))) __bf16 v16bf;
typedef __attribute__((ext_vector_type(8)))  __bf16 v8bf;
typedef __attribute__((ext_vector_type(8)))  float  v8f;
typedef __attribute__((ext_vector_type(4)))  float  v4f;
typedef __attribute__((ext_vector_type(4)))  unsigned int u32x4;
typedef __attribute__((ext_vector_type(8)))  int i32x8;
typedef __attribute__((ext_vector_type(4)))  int i32x4;

// LDS row stride for TDM-staged weight strips: 128 dwords data + 4 dwords pad
// (pad breaks the 512B row period so the 16 column-lanes of a WMMA B-fragment
// hit distinct LDS banks: bank advance = 132 % 64 = 4 dwords/row).
#define WROW_BF16 264   // 132 dwords = 528 bytes per 64-col strip row
#define WSTRIP_BF16 (64 * WROW_BF16)

__device__ __forceinline__ v8f wmma_bf16(v16bf a, v16bf b, v8f c) {
  return __builtin_amdgcn_wmma_f32_16x16x32_bf16(
      /*neg_a=*/false, a, /*neg_b=*/false, b,
      /*c_mod=*/(short)0, c, /*reuse_a=*/false, /*reuse_b=*/false);
}

// ---------------------------------------------------------------------------
// TDM: DMA a 64-row x 512B weight strip (one 64-column group of a transposed
// 256x256 bf16 weight matrix, contiguous 32KB in global) into LDS, inserting
// 4 dwords of padding after every 128-dword row (ISA 08_async_tensor §8).
// 2D tile: data_size=4B, tensor_dim0=tile_dim0=128 dw, tensor_dim1=tile_dim1=64,
// dim0 stride=128 dw, pad_enable, pad_interval=128 dw (code 6), pad_amount=4 dw
// (code 3). Descriptor group0 word3 carries global_addr[56:32] | type=2.
// ---------------------------------------------------------------------------
__device__ __forceinline__ void tdm_load_strip(const __bf16* gsrc,
                                               unsigned lds_byte_off) {
  unsigned long long ga = (unsigned long long)(uintptr_t)gsrc;
  u32x4 g0;
  g0[0] = 1u;                                   // count=1, user descriptor
  g0[1] = lds_byte_off;                         // lds_addr
  g0[2] = (unsigned)(ga & 0xffffffffu);         // global_addr[31:0]
  g0[3] = (unsigned)((ga >> 32) & 0x01ffffffu)  // global_addr[56:32]
          | (2u << 30);                         // type=2 ("image")
  i32x8 g1;
  g1[0] = (2 << 16)      // data_size = 4B
        | (1 << 20)      // pad_enable
        | (6 << 22)      // pad_interval: 128 dwords
        | (3 << 25);     // pad_amount:   4 dwords
  g1[1] = (128 << 16);   // tensor_dim0[15:0] = 128 (bits 48-63)
  g1[2] = (64 << 16);    // tensor_dim1[15:0] = 64  (bits 80-95)
  g1[3] = (128 << 16);   // tile_dim0 = 128         (bits 112-127)
  g1[4] = 64;            // tile_dim1 = 64, tile_dim2 = 0
  g1[5] = 128;           // tensor_dim0_stride[31:0] = 128 dwords (dense)
  g1[6] = 0;
  g1[7] = 0;
  i32x4 gz = {0, 0, 0, 0};
#if defined(__clang_major__) && (__clang_major__ >= 23)
  i32x8 gz8 = {0, 0, 0, 0, 0, 0, 0, 0};
  __builtin_amdgcn_tensor_load_to_lds(g0, g1, gz, gz, gz8, 0);
#else
  __builtin_amdgcn_tensor_load_to_lds(g0, g1, gz, gz, 0);
#endif
}

// Load a WMMA B-fragment from a TDM-staged LDS strip.
// Lane holds column `colInStrip`, K = k0 + 16*hi + [0..15] contiguous bf16.
__device__ __forceinline__ v16bf ld_bfrag_lds(const __bf16* p) {
  v8bf b0 = *(const v8bf*)(p);
  v8bf b1 = *(const v8bf*)(p + 8);
  v16bf r;
#pragma unroll
  for (int e = 0; e < 8; ++e) { r[e] = b0[e]; r[e + 8] = b1[e]; }
  return r;
}

// ---------------------------------------------------------------------------
// Kernel 0: transpose + bf16-convert the three 256x256 weight matrices.
// W is row-major [k][n]; WT is [n][k] so B-fragments are contiguous in K.
// ---------------------------------------------------------------------------
__global__ void rm_prep_weights(const float* __restrict__ Wq,
                                const float* __restrict__ Wo,
                                const float* __restrict__ Wc,
                                __bf16* __restrict__ WqT,
                                __bf16* __restrict__ WoT,
                                __bf16* __restrict__ WcT) {
  const int idx = blockIdx.x * blockDim.x + threadIdx.x;  // k*256 + n
  const int k = idx >> 8;
  const int n = idx & 255;
  const int tdx = n * Dsz + k;
  WqT[tdx] = (__bf16)Wq[idx];
  WoT[tdx] = (__bf16)Wo[idx];
  WcT[tdx] = (__bf16)Wc[idx];
}

// ---------------------------------------------------------------------------
// Kernel 1: fused Q/C projection GEMM.
//   Qb = bf16(H @ Wq + bq),  Cb = bf16(H @ Wc + bc)
// Block = 8 row-tiles x one 64-column group. Wave 0 TDM-loads the Wq and Wc
// strips (2 x 33KB, padded) into LDS once; all 8 waves then feed WMMA B
// operands from LDS while streaming the A-matrix (f32 H) from global.
// ---------------------------------------------------------------------------
__global__ __launch_bounds__(256, 2)
void rm_gemm_qc(const float* __restrict__ H,
                const __bf16* __restrict__ WqT,
                const __bf16* __restrict__ WcT,
                const float* __restrict__ bq,
                const float* __restrict__ bc,
                __bf16* __restrict__ Qb,
                __bf16* __restrict__ Cb) {
  __shared__ __align__(32) __bf16 wsh[2 * WSTRIP_BF16];   // Wq strip | Wc strip

  const int lane = threadIdx.x & 31;
  const int wave = threadIdx.x >> 5;
  const int mblk = blockIdx.x >> 2;       // row-tile block
  const int nb   = blockIdx.x & 3;        // 64-column group
  const int m0    = (mblk * 8 + wave) * 16;
  const int nbase = nb * 64;
  const int r  = lane & 15;
  const int hi = lane >> 4;

  if (wave == 0) {
    const unsigned lds0 = (unsigned)(uintptr_t)(void*)&wsh[0];
    tdm_load_strip(WqT + (size_t)nbase * Dsz, lds0);
    tdm_load_strip(WcT + (size_t)nbase * Dsz, lds0 + WSTRIP_BF16 * 2u);
    __builtin_amdgcn_s_wait_tensorcnt((short)0);
  }
  __syncthreads();

  const float* arow = H + (size_t)(m0 + r) * Dsz;
  const __bf16* wq_l = wsh;
  const __bf16* wc_l = wsh + WSTRIP_BF16;

  v8f z = {};
  v8f accq[4], accc[4];
#pragma unroll
  for (int g = 0; g < 4; ++g) { accq[g] = z; accc[g] = z; }

  for (int k0 = 0; k0 < Dsz; k0 += 32) {
    // A fragment: lane holds K = 16*hi + [0..15]
    const float* ap = arow + k0 + 16 * hi;
    v4f a0 = *(const v4f*)(ap);
    v4f a1 = *(const v4f*)(ap + 4);
    v4f a2 = *(const v4f*)(ap + 8);
    v4f a3 = *(const v4f*)(ap + 12);
    v16bf a;
#pragma unroll
    for (int e = 0; e < 4; ++e) {
      a[e]      = (__bf16)a0[e];
      a[e + 4]  = (__bf16)a1[e];
      a[e + 8]  = (__bf16)a2[e];
      a[e + 12] = (__bf16)a3[e];
    }
#pragma unroll
    for (int g = 0; g < 4; ++g) {
      const int strip_off = (g * 16 + r) * WROW_BF16 + k0 + 16 * hi;
      v16bf bqf = ld_bfrag_lds(wq_l + strip_off);
      v16bf bcf = ld_bfrag_lds(wc_l + strip_off);
      accq[g] = wmma_bf16(a, bqf, accq[g]);
      accc[g] = wmma_bf16(a, bcf, accc[g]);
    }
  }

#pragma unroll
  for (int g = 0; g < 4; ++g) {
    const int col = nbase + g * 16 + r;
    const float vbq = bq[col];
    const float vbc = bc[col];
#pragma unroll
    for (int i = 0; i < 8; ++i) {
      const size_t o = (size_t)(m0 + i + 8 * hi) * Dsz + col;
      Qb[o] = (__bf16)(accq[g][i] + vbq);
      Cb[o] = (__bf16)(accc[g][i] + vbc);
    }
  }
}

// ---------------------------------------------------------------------------
// Kernel 2: sequential ring-buffer scan. One workgroup per batch (128 WGs),
// 256 threads, buffer resident in LDS as bf16 (32 KB).
// ---------------------------------------------------------------------------
__global__ __launch_bounds__(256)
void rm_ring_seq(const float* __restrict__ masks,
                 const float* __restrict__ buffer0,
                 const __bf16* __restrict__ Qb,
                 const __bf16* __restrict__ Cb,
                 __bf16* __restrict__ Rb,
                 float* __restrict__ out_buf,
                 float* __restrict__ out_ptr) {
  __shared__ __bf16 buf[Nsz * Dsz];     // 32 KB ring buffer
  __shared__ float  qsh[Dsz];
  __shared__ float  part[Nsz * 4];      // quarter-row dot partials
  __shared__ float  lg[Nsz];            // logits
  __shared__ float  aw[Nsz];            // normalized attention weights

  const int b   = blockIdx.x;
  const int tid = threadIdx.x;

  for (int i = tid; i < Nsz * Dsz; i += 256)
    buf[i] = (__bf16)buffer0[(size_t)b * Nsz * Dsz + i];
  int ptr = 0;
  __syncthreads();

  for (int t = 0; t < Tsz; ++t) {
    const size_t base = ((size_t)b * Tsz + t) * Dsz;
    const float m = masks[(size_t)b * Tsz + t];
    if (m != 1.0f) {                    // episode reset / rescale
      for (int i = tid; i < Nsz * Dsz; i += 256)
        buf[i] = (__bf16)((float)buf[i] * m);
      ptr = ptr * (int)m;
      __syncthreads();
    }

    qsh[tid] = (float)Qb[base + tid];
    __syncthreads();

    // logits: slot n = tid>>2 handles quarter (tid&3) of D
    {
      const int n  = tid >> 2;
      const int q0 = (tid & 3) * 64;
      const __bf16* brow = buf + n * Dsz + q0;
      const float*  qp   = qsh + q0;
      float s = 0.f;
#pragma unroll 8
      for (int j = 0; j < 64; ++j) s += (float)brow[j] * qp[j];
      part[tid] = s;                    // part[n*4 + quarter] == part[tid]
    }
    __syncthreads();

    if (tid < Nsz)
      lg[tid] = (part[tid * 4] + part[tid * 4 + 1] +
                 part[tid * 4 + 2] + part[tid * 4 + 3]) * 0.0625f;  // 1/sqrt(D)
    __syncthreads();

    // softmax stats computed redundantly per thread (no extra barriers)
    float mx = -3.0e38f;
    for (int n2 = 0; n2 < Nsz; ++n2) mx = fmaxf(mx, lg[n2]);
    float se = 0.f;
    for (int n2 = 0; n2 < Nsz; ++n2) se += __expf(lg[n2] - mx);
    if (tid < Nsz) aw[tid] = __expf(lg[tid] - mx) / se;
    __syncthreads();

    // readvec (thread owns column tid), then scatter comp into slot ptr
    float rv = 0.f;
#pragma unroll 8
    for (int n2 = 0; n2 < Nsz; ++n2) rv += aw[n2] * (float)buf[n2 * Dsz + tid];
    Rb[base + tid] = (__bf16)rv;
    buf[ptr * Dsz + tid] = Cb[base + tid];
    ptr = (ptr + 1) & (Nsz - 1);
    __syncthreads();
  }

  for (int i = tid; i < Nsz * Dsz; i += 256)
    out_buf[(size_t)b * Nsz * Dsz + i] = (float)buf[i];
  if (tid == 0) out_ptr[b] = (float)ptr;
}

// ---------------------------------------------------------------------------
// Kernel 3: output projection GEMM.  read_seq = f32(Rb @ Wo + bo)
// Same TDM-staged weight strip scheme as kernel 1, single weight matrix.
// ---------------------------------------------------------------------------
__global__ __launch_bounds__(256, 2)
void rm_gemm_out(const __bf16* __restrict__ Rb,
                 const __bf16* __restrict__ WoT,
                 const float* __restrict__ bo,
                 float* __restrict__ Out) {
  __shared__ __align__(32) __bf16 wsh[WSTRIP_BF16];

  const int lane = threadIdx.x & 31;
  const int wave = threadIdx.x >> 5;
  const int mblk = blockIdx.x >> 2;
  const int nb   = blockIdx.x & 3;
  const int m0    = (mblk * 8 + wave) * 16;
  const int nbase = nb * 64;
  const int r  = lane & 15;
  const int hi = lane >> 4;

  if (wave == 0) {
    tdm_load_strip(WoT + (size_t)nbase * Dsz,
                   (unsigned)(uintptr_t)(void*)&wsh[0]);
    __builtin_amdgcn_s_wait_tensorcnt((short)0);
  }
  __syncthreads();

  const __bf16* arow = Rb + (size_t)(m0 + r) * Dsz;

  v8f z = {};
  v8f acc[4];
#pragma unroll
  for (int g = 0; g < 4; ++g) acc[g] = z;

  for (int k0 = 0; k0 < Dsz; k0 += 32) {
    const __bf16* ap = arow + k0 + 16 * hi;
    v8bf alo = *(const v8bf*)(ap);
    v8bf ahi = *(const v8bf*)(ap + 8);
    v16bf a;
#pragma unroll
    for (int e = 0; e < 8; ++e) { a[e] = alo[e]; a[e + 8] = ahi[e]; }
#pragma unroll
    for (int g = 0; g < 4; ++g) {
      const int strip_off = (g * 16 + r) * WROW_BF16 + k0 + 16 * hi;
      acc[g] = wmma_bf16(a, ld_bfrag_lds(wsh + strip_off), acc[g]);
    }
  }

#pragma unroll
  for (int g = 0; g < 4; ++g) {
    const int col = nbase + g * 16 + r;
    const float vb = bo[col];
#pragma unroll
    for (int i = 0; i < 8; ++i)
      Out[(size_t)(m0 + i + 8 * hi) * Dsz + col] = acc[g][i] + vb;
  }
}

// ---------------------------------------------------------------------------
extern "C" void kernel_launch(void* const* d_in, const int* in_sizes, int n_in,
                              void* d_out, int out_size, void* d_ws, size_t ws_size,
                              hipStream_t stream) {
  (void)in_sizes; (void)n_in; (void)out_size; (void)ws_size;

  const float* H       = (const float*)d_in[0];  // hidden_seq (B,T,D)
  const float* masks   = (const float*)d_in[1];  // (B,T)
  const float* buffer0 = (const float*)d_in[2];  // (B,N,D)
  const float* Wq      = (const float*)d_in[3];
  const float* bq      = (const float*)d_in[4];
  const float* Wo      = (const float*)d_in[5];
  const float* bo      = (const float*)d_in[6];
  const float* Wc      = (const float*)d_in[7];
  const float* bc      = (const float*)d_in[8];

  const size_t RT = (size_t)Bsz * Tsz;           // 131072 token rows

  // Workspace layout: Qb | Cb | Rb (bf16, 64 MB each) | WqT | WoT | WcT
  char* ws = (char*)d_ws;
  __bf16* Qb  = (__bf16*)ws;
  __bf16* Cb  = (__bf16*)(ws + 1 * RT * Dsz * sizeof(__bf16));
  __bf16* Rb  = (__bf16*)(ws + 2 * RT * Dsz * sizeof(__bf16));
  __bf16* WqT = (__bf16*)(ws + 3 * RT * Dsz * sizeof(__bf16));
  __bf16* WoT = WqT + Dsz * Dsz;
  __bf16* WcT = WoT + Dsz * Dsz;

  // Output tuple laid out flat: read_seq (B,T,D) | buf_f (B,N,D) | ptr_f (B,)
  float* out_read = (float*)d_out;
  float* out_buf  = out_read + RT * Dsz;
  float* out_ptr  = out_buf + (size_t)Bsz * Nsz * Dsz;

  // GEMM grid: (RT/16 row-tiles / 8 per block) x 4 column groups
  const int gemm_blocks = (int)(RT / 16 / 8) * (Dsz / 64);  // 4096

  rm_prep_weights<<<(Dsz * Dsz) / 256, 256, 0, stream>>>(Wq, Wo, Wc, WqT, WoT, WcT);
  rm_gemm_qc<<<gemm_blocks, 256, 0, stream>>>(H, WqT, WcT, bq, bc, Qb, Cb);
  rm_ring_seq<<<Bsz, 256, 0, stream>>>(masks, buffer0, Qb, Cb, Rb, out_buf, out_ptr);
  rm_gemm_out<<<gemm_blocks, 256, 0, stream>>>(Rb, WoT, bo, out_read);
}
